// Decoder_56513179681273
// MI455X (gfx1250) — matrix operand order repaired and checked by reading, not compiled, not run on hardware
//
#include <hip/hip_runtime.h>
#include <math.h>

// Sizes fixed by the reference
#define BB 512
#define TT 512
#define HH 128

typedef __attribute__((ext_vector_type(16))) _Float16 v16h;
typedef __attribute__((ext_vector_type(8)))  float    v8f;

static __device__ __forceinline__ unsigned short cvt_f16(float f) {
  return __builtin_bit_cast(unsigned short, (_Float16)f);   // v_cvt_f16_f32
}
static __device__ __forceinline__ unsigned pack_f16(float f0, float f1) {
  return (unsigned)cvt_f16(f0) | ((unsigned)cvt_f16(f1) << 16);
}
static __device__ __forceinline__ float h2f(unsigned short h) {
  return (float)__builtin_bit_cast(_Float16, h);            // v_cvt_f32_f16
}
static __device__ __forceinline__ _Float16 us2h(unsigned short h) {
  return __builtin_bit_cast(_Float16, h);
}
static __device__ __forceinline__ float tanh_fast(float x) {
#if __has_builtin(__builtin_amdgcn_tanhf)
  return __builtin_amdgcn_tanhf(x);                         // gfx1250 v_tanh_f32 (TRANS)
#else
  float e = __expf(-2.0f * fabsf(x));
  float t = (1.0f - e) / (1.0f + e);
  return copysignf(t, x);
#endif
}
static __device__ __forceinline__ float sigm(float x) {
#if __has_builtin(__builtin_amdgcn_tanhf)
  return __builtin_fmaf(0.5f, __builtin_amdgcn_tanhf(0.5f * x), 0.5f);
#else
  return 1.0f / (1.0f + __expf(-x));
#endif
}

// ---- WMMA fragment loaders (ISA 7.12.2 layouts, wave32) ----

// A fragment 16x32 f16 from LDS holding h as packed f16 pairs, row pitch 64 u32 (K=128).
static __device__ __forceinline__ v16h load_a_pairs(const unsigned int* src, int lane, int ks) {
  v16h a;
  int row  = lane & 15;
  int half = (lane < 16) ? 0 : 4;          // pair units (8 K values / 2)
  const unsigned int* r = src + row * 64 + ks * 16;
#pragma unroll
  for (int j = 0; j < 4; ++j) {
    unsigned int u = r[half + j];
    a[2*j]   = us2h((unsigned short)(u & 0xFFFFu));
    a[2*j+1] = us2h((unsigned short)(u >> 16));
  }
#pragma unroll
  for (int j = 0; j < 4; ++j) {
    unsigned int u = r[8 + half + j];
    a[8+2*j]   = us2h((unsigned short)(u & 0xFFFFu));
    a[8+2*j+1] = us2h((unsigned short)(u >> 16));
  }
  return a;
}

// A fragment 16x32 from global fp32 rows (pitch 128 floats), converted to f16.
static __device__ __forceinline__ v16h load_a_f32(const float* src, int rowbase, int lane, int ks) {
  v16h a;
  int row  = rowbase + (lane & 15);
  int half = (lane < 16) ? 0 : 4;
  const float2* r = (const float2*)(src + (size_t)row * 128) + ks * 16;
#pragma unroll
  for (int j = 0; j < 4; ++j) {
    float2 u = r[half + j];
    a[2*j]   = (_Float16)u.x;
    a[2*j+1] = (_Float16)u.y;
  }
#pragma unroll
  for (int j = 0; j < 4; ++j) {
    float2 u = r[8 + half + j];
    a[8+2*j]   = (_Float16)u.x;
    a[8+2*j+1] = (_Float16)u.y;
  }
  return a;
}

// B fragment 32x16 f16 from a [N][K] (K-contiguous) f16 matrix, row pitch 64 u32 (K=128).
static __device__ __forceinline__ v16h load_b_t(const unsigned int* srcT, int ntile, int lane, int ks) {
  v16h b;
  int n = ntile * 16 + (lane & 15);
  const unsigned int* r = srcT + n * 64 + ks * 16 + ((lane < 16) ? 0 : 4);
  uint4 q0 = *(const uint4*)(r);
  uint4 q1 = *(const uint4*)(r + 8);
  unsigned w0[4] = {q0.x, q0.y, q0.z, q0.w};
  unsigned w1[4] = {q1.x, q1.y, q1.z, q1.w};
#pragma unroll
  for (int j = 0; j < 4; ++j) {
    b[2*j]     = us2h((unsigned short)(w0[j] & 0xFFFFu));
    b[2*j+1]   = us2h((unsigned short)(w0[j] >> 16));
    b[8+2*j]   = us2h((unsigned short)(w1[j] & 0xFFFFu));
    b[8+2*j+1] = us2h((unsigned short)(w1[j] >> 16));
  }
  return b;
}

// ---- Kernel 1: transpose + convert weights to f16 [N][K] in workspace ----
__global__ __launch_bounds__(256) void prep_kernel(
    const float* __restrict__ rec, const float* __restrict__ W1, const float* __restrict__ W2,
    unsigned short* __restrict__ recT, unsigned short* __restrict__ W1T,
    unsigned short* __restrict__ W2T) {
  int idx = blockIdx.x * 256 + threadIdx.x;
  const int nRec = 512 * 128, nW = 128 * 128;
  if (idx < nRec) {
    int n = idx / 128, k = idx % 128;
    recT[n * 128 + k] = cvt_f16(rec[k * 512 + n]);
  } else if (idx < nRec + nW) {
    int i = idx - nRec; int n = i / 128, k = i % 128;
    W1T[n * 128 + k] = cvt_f16(W1[k * 128 + n]);
  } else if (idx < nRec + 2 * nW) {
    int i = idx - nRec - nW; int n = i / 128, k = i % 128;
    W2T[n * 128 + k] = cvt_f16(W2[k * 128 + n]);
  }
}

// ---- Kernel 2: w1e = enc_output @ W1, f16 WMMA, output f16 [B*T][H] ----
__global__ __launch_bounds__(256) void w1e_kernel(
    const float* __restrict__ enc, const unsigned int* __restrict__ W1T,
    unsigned short* __restrict__ w1e) {
  int wave = threadIdx.x >> 5, lane = threadIdx.x & 31;
  int strip = blockIdx.x * 8 + wave;          // 16-row strip of (B*T) rows
  int rowbase = strip * 16;
  int mbase = (lane < 16) ? 0 : 8;
#pragma unroll 1
  for (int nt = 0; nt < 8; ++nt) {
    v8f c = {};
#pragma unroll
    for (int ks = 0; ks < 4; ++ks) {
      v16h a = load_a_f32(enc, rowbase, lane, ks);
      v16h b = load_b_t(W1T, nt, lane, ks);
      c = __builtin_amdgcn_wmma_f32_16x16x32_f16(false, a, false, b, (short)0, c, false, false);
    }
    int col = nt * 16 + (lane & 15);
#pragma unroll
    for (int e = 0; e < 8; ++e)
      w1e[(size_t)(rowbase + mbase + e) * 128 + col] = cvt_f16(c[e]);
  }
}

// ---- Kernel 3: persistent decoder; block = 16 batch rows, full 512-step scan ----
__global__ __launch_bounds__(256) void decoder_kernel(
    const float* __restrict__ x, const float* __restrict__ h0, const float* __restrict__ c0,
    const float* __restrict__ kern, const float* __restrict__ bias,
    const unsigned int* __restrict__ recT, const unsigned int* __restrict__ W2T,
    const unsigned short* __restrict__ w1e, const float* __restrict__ V,
    float* __restrict__ out) {
  __shared__ float        zbuf[16 * 512];   // z gates, fp32
  __shared__ float        cbuf[16 * 128];   // cell state
  __shared__ unsigned int hbf [16 * 64];    // h as packed f16 pairs (K-contig)
  __shared__ float        w2h [16 * 128];   // h2 @ W2
  __shared__ float        sK0[512], sK1[512], sBias[512];
  __shared__ float        sV[128];
  __shared__ float        sPtr[16][2];
  __shared__ float        red_val[256];
  __shared__ int          red_idx[256];

  int tid  = threadIdx.x;
  int wave = tid >> 5, lane = tid & 31;
  int mbase = (lane < 16) ? 0 : 8;
  int bg = blockIdx.x * 16;

  // ---- init ----
  for (int j = tid; j < 512; j += 256) { sK0[j] = kern[j]; sK1[j] = kern[512 + j]; sBias[j] = bias[j]; }
  if (tid < 128) sV[tid] = V[tid];
  if (tid < 32)  sPtr[tid >> 1][tid & 1] = 1.0f;
  for (int n = tid; n < 16 * 128; n += 256) {
    int b = n >> 7, j = n & 127;
    cbuf[n] = c0[(size_t)(bg + b) * 128 + j];
  }
  for (int n = tid; n < 16 * 64; n += 256) {
    int b = n >> 6, jp = n & 63;
    float f0 = h0[(size_t)(bg + b) * 128 + 2 * jp];
    float f1 = h0[(size_t)(bg + b) * 128 + 2 * jp + 1];
    hbf[n] = pack_f16(f0, f1);
  }
  __syncthreads();

  for (int step = 0; step < TT; ++step) {
    // ---- z = ptr@kernel + bias + h@rec_kernel  (WMMA, 32 N-tiles over 8 waves) ----
#pragma unroll 1
    for (int q = 0; q < 4; ++q) {
      int nt  = wave * 4 + q;
      int col = nt * 16 + (lane & 15);
      v8f c;
#pragma unroll
      for (int e = 0; e < 8; ++e) {
        int m = mbase + e;
        c[e] = sBias[col] + sPtr[m][0] * sK0[col] + sPtr[m][1] * sK1[col];
      }
#pragma unroll
      for (int ks = 0; ks < 4; ++ks) {
        v16h a = load_a_pairs(hbf, lane, ks);
        v16h b = load_b_t(recT, nt, lane, ks);
        c = __builtin_amdgcn_wmma_f32_16x16x32_f16(false, a, false, b, (short)0, c, false, false);
      }
#pragma unroll
      for (int e = 0; e < 8; ++e) zbuf[(mbase + e) * 512 + col] = c[e];
    }
    __syncthreads();

    // ---- LSTM gates (i,f,g,o) -> c2, h2 (h2 stored as packed f16) ----
    for (int n = tid; n < 16 * 64; n += 256) {
      int b = n >> 6, jp = n & 63;
      float hv[2];
#pragma unroll
      for (int s = 0; s < 2; ++s) {
        int j = 2 * jp + s;
        float zi = zbuf[b * 512 + j];
        float zf = zbuf[b * 512 + 128 + j];
        float zg = zbuf[b * 512 + 256 + j];
        float zo = zbuf[b * 512 + 384 + j];
        float cc = sigm(zf) * cbuf[b * 128 + j] + sigm(zi) * tanh_fast(zg);
        cbuf[b * 128 + j] = cc;
        hv[s] = sigm(zo) * tanh_fast(cc);
      }
      hbf[n] = pack_f16(hv[0], hv[1]);
    }
    __syncthreads();

    // ---- w2h = h2 @ W2  (8 N-tiles, one per wave) ----
    {
      int nt = wave;
      v8f c = {};
#pragma unroll
      for (int ks = 0; ks < 4; ++ks) {
        v16h a = load_a_pairs(hbf, lane, ks);
        v16h b = load_b_t(W2T, nt, lane, ks);
        c = __builtin_amdgcn_wmma_f32_16x16x32_f16(false, a, false, b, (short)0, c, false, false);
      }
      int col = nt * 16 + (lane & 15);
#pragma unroll
      for (int e = 0; e < 8; ++e) w2h[(mbase + e) * 128 + col] = c[e];
    }
    __syncthreads();

    // ---- scores[b,t] = V . tanh(w1e[b,t,:] + w2h[b,:]); write out; local argmax ----
    {
      int b = tid >> 4;          // 16 threads per batch row
      int tl = tid & 15;
      const unsigned short* wrow = w1e + (size_t)(bg + b) * 512 * 128;
      float* orow = out + ((size_t)(bg + b) * 512 + step) * 512;
      float bestv = -__builtin_inff(); int besti = 0;
      for (int t = tl; t < 512; t += 16) {
        const uint4* wr = (const uint4*)(wrow + (size_t)t * 128);
        if (t + 16 < 512)   // prefetch next row for this thread (global_prefetch_b8)
          __builtin_prefetch((const void*)(wrow + (size_t)(t + 16) * 128), 0, 1);
        float acc = 0.0f;
#pragma unroll 4
        for (int hq = 0; hq < 16; ++hq) {       // 16 * 8 f16 = 128
          uint4 u = wr[hq];
          unsigned w[4] = {u.x, u.y, u.z, u.w};
          int h = hq * 8;
#pragma unroll
          for (int d = 0; d < 4; ++d) {
            float e0 = h2f((unsigned short)(w[d] & 0xFFFFu));
            float e1 = h2f((unsigned short)(w[d] >> 16));
            float t0 = tanh_fast(e0 + w2h[b * 128 + h + 2 * d]);
            float t1 = tanh_fast(e1 + w2h[b * 128 + h + 2 * d + 1]);
            acc += t0 * sV[h + 2 * d] + t1 * sV[h + 2 * d + 1];
          }
        }
        orow[t] = acc;
        if (acc > bestv) { bestv = acc; besti = t; }
      }
      red_val[tid] = bestv; red_idx[tid] = besti;
    }
    __syncthreads();

    // ---- argmax reduce within each 16-thread group; gather next ptr from x ----
#pragma unroll
    for (int off = 8; off >= 1; off >>= 1) {
      if ((tid & 15) < off) {
        float v2 = red_val[tid + off]; int i2 = red_idx[tid + off];
        if (v2 > red_val[tid] || (v2 == red_val[tid] && i2 < red_idx[tid])) {
          red_val[tid] = v2; red_idx[tid] = i2;
        }
      }
      __syncthreads();
    }
    if ((tid & 15) == 0) {
      int b = tid >> 4;
      int idx = red_idx[tid];
      const float* xp = x + ((size_t)(bg + b) * 512 + idx) * 2;
      sPtr[b][0] = xp[0]; sPtr[b][1] = xp[1];
    }
    __syncthreads();
  }
}

extern "C" void kernel_launch(void* const* d_in, const int* in_sizes, int n_in,
                              void* d_out, int out_size, void* d_ws, size_t ws_size,
                              hipStream_t stream) {
  (void)in_sizes; (void)n_in; (void)out_size; (void)ws_size;
  const float* x    = (const float*)d_in[0];
  const float* enc  = (const float*)d_in[1];
  const float* h0   = (const float*)d_in[2];
  const float* c0   = (const float*)d_in[3];
  const float* kern = (const float*)d_in[4];
  const float* rec  = (const float*)d_in[5];
  const float* bias = (const float*)d_in[6];
  const float* W1   = (const float*)d_in[7];
  const float* W2   = (const float*)d_in[8];
  const float* V    = (const float*)d_in[9];
  float* out = (float*)d_out;

  // workspace layout (f16): w1e [B*T*H], recT [512*128], W1T [128*128], W2T [128*128]
  char* ws = (char*)d_ws;
  unsigned short* w1e  = (unsigned short*)ws;
  unsigned short* recT = (unsigned short*)(ws + (size_t)BB * TT * HH * 2);
  unsigned short* W1T  = recT + 512 * 128;
  unsigned short* W2T  = W1T + 128 * 128;

  // 1) weight transpose/convert: (512*128 + 2*128*128) elements / 256 per block
  prep_kernel<<<384, 256, 0, stream>>>(rec, W1, W2, recT, W1T, W2T);
  // 2) w1e precompute: 262144 rows -> 16384 strips -> 2048 blocks * 8 waves
  w1e_kernel<<<2048, 256, 0, stream>>>(enc, (const unsigned int*)W1T, w1e);
  // 3) persistent decoder: 32 blocks * 16 batch rows
  decoder_kernel<<<32, 256, 0, stream>>>(x, h0, c0, kern, bias,
                                         (const unsigned int*)recT,
                                         (const unsigned int*)W2T,
                                         w1e, V, out);
}
